// MambaSemanticMapper_50208167690327
// MI455X (gfx1250) — matrix operand order repaired and checked by reading, not compiled, not run on hardware
//
#include <hip/hip_runtime.h>
#include <hip/hip_bf16.h>
#include <math.h>

// ---------------- problem constants ----------------
#define BATCH   256
#define SEQL    77
#define DMODEL  768
#define HID     256
#define DI      512        // D_INNER
#define DST     16         // D_STATE
#define DTR     16         // DT_RANK
#define NTOK    (BATCH * SEQL)   // 19712 (multiple of 64 -> no row checks)
#define XDBLW   (DTR + 2 * DST)  // 48

// ---------------- WMMA types ----------------
typedef __attribute__((ext_vector_type(16))) __bf16 v16bf;
typedef __attribute__((ext_vector_type(8)))  float  v8f;

union Frag {
  v16bf v;
  uint4 q[2];
};

__device__ __forceinline__ unsigned f2bfu(float f) {
  unsigned u = __float_as_uint(f);
  unsigned r = u + 0x7FFFu + ((u >> 16) & 1u);   // round-to-nearest-even
  return r >> 16;
}
__device__ __forceinline__ unsigned packbf2(float lo, float hi) {
  return f2bfu(lo) | (f2bfu(hi) << 16);
}
__device__ __forceinline__ uint4 packbf8(float4 a, float4 b) {
  return make_uint4(packbf2(a.x, a.y), packbf2(a.z, a.w),
                    packbf2(b.x, b.y), packbf2(b.z, b.w));
}
__device__ __forceinline__ float sigmoidf_(float v) {
  return 1.f / (1.f + __expf(-v));
}

// ---------------------------------------------------------------------------
// GEMM:  C[N,M] = act( X[N,K(ldx)] * W[M,K]^T + bias )      bf16 WMMA
// Block: 256 threads = 8 waves (2x4). Block tile 64(N) x 128(M).
// Each wave: 2x2 subtiles of 16x16 -> 4 WMMA per k-step, frag reuse 2x.
// Assumes N % 64 == 0 and K % 16 == 0 (true for every launch here).
// CK : K may not be a multiple of 32 (zero-pad).
// CM : M may not cover the whole 128-col tile (mask columns).
// ACT: 1 -> softplus epilogue.     ACC: accumulate into C.
// ---------------------------------------------------------------------------
template <bool CK, bool CM, int ACT, bool ACC>
__global__ __launch_bounds__(256) void gemm_bf16_wmma_kernel(
    const float* __restrict__ X, int ldx,
    const float* __restrict__ W,
    const float* __restrict__ bias,
    float* __restrict__ C,
    int N, int K, int M) {
  // row stride 40 shorts = 80B: 16B-aligned rows, conflict-free b128 reads
  __shared__ unsigned short As[64][40];    // [row][k]
  __shared__ unsigned short Bs[128][40];   // [m(col)][k]

  const int tid     = threadIdx.x;
  const int rowBase = blockIdx.x * 64;
  const int mBase   = blockIdx.y * 128;

  const int lane   = tid & 31;
  const int waveId = tid >> 5;
  const int wr     = waveId >> 2;   // 0..1  -> 32-row slice
  const int wc     = waveId & 3;    // 0..3  -> 32-col slice
  const int half   = lane >> 4;     // half-wave selector
  const int lid    = lane & 15;

  // staging indices: A = 64x32 (8 floats/thread), B = 128x32 (16 floats/thread)
  const int ar  = tid >> 2;          // 0..63
  const int akc = (tid & 3) << 3;    // 0,8,16,24
  const int bm  = tid >> 1;          // 0..127
  const int bkc = (tid & 1) << 4;    // 0,16

  const float* pA = X + (size_t)(rowBase + ar) * ldx + akc;
  const int    gm = mBase + bm;
  const float* pB = W + (size_t)gm * K + bkc;
  const bool   bmOK = !CM || (gm < M);

  v8f a00 = {0.f,0.f,0.f,0.f,0.f,0.f,0.f,0.f};
  v8f a01 = a00, a10 = a00, a11 = a00;

  for (int k0 = 0; k0 < K; k0 += 32) {
    // ---- stage A (two float4 -> one b128 LDS store) ----
    float4 av0 = make_float4(0.f,0.f,0.f,0.f), av1 = av0;
    if (!CK || (k0 + akc) < K)     av0 = *(const float4*)(pA);
    if (!CK || (k0 + akc + 4) < K) av1 = *(const float4*)(pA + 4);
    *(uint4*)&As[ar][akc] = packbf8(av0, av1);

    // ---- stage B (four float4 -> two b128 LDS stores) ----
    float4 bv0 = make_float4(0.f,0.f,0.f,0.f), bv1 = bv0, bv2 = bv0, bv3 = bv0;
    if (bmOK) {
      if (!CK || (k0 + bkc) < K)      bv0 = *(const float4*)(pB);
      if (!CK || (k0 + bkc + 4) < K)  bv1 = *(const float4*)(pB + 4);
      if (!CK || (k0 + bkc + 8) < K)  bv2 = *(const float4*)(pB + 8);
      if (!CK || (k0 + bkc + 12) < K) bv3 = *(const float4*)(pB + 12);
    }
    *(uint4*)&Bs[bm][bkc]     = packbf8(bv0, bv1);
    *(uint4*)&Bs[bm][bkc + 8] = packbf8(bv2, bv3);
    pA += 32;
    pB += 32;
    __syncthreads();

    // ---- fragments: half h holds k 8h..8h+7, 16+8h..16+8h+7 -> 2x ds_load_b128 each ----
    Frag fa0, fa1, fb0, fb1;
    const int ar0 = wr * 32 + lid, ar1 = ar0 + 16;
    const int bc0 = wc * 32 + lid, bc1 = bc0 + 16;
    const int ha = half * 8, hb = half * 16;
    fa0.q[0] = *(const uint4*)&As[ar0][ha];
    fa0.q[1] = *(const uint4*)&As[ar0][16 + ha];
    fa1.q[0] = *(const uint4*)&As[ar1][ha];
    fa1.q[1] = *(const uint4*)&As[ar1][16 + ha];
    fb0.q[0] = *(const uint4*)&Bs[bc0][ha];
    fb0.q[1] = *(const uint4*)&Bs[bc0][16 + ha];
    fb1.q[0] = *(const uint4*)&Bs[bc1][ha];
    fb1.q[1] = *(const uint4*)&Bs[bc1][16 + ha];

    a00 = __builtin_amdgcn_wmma_f32_16x16x32_bf16(false, fa0.v, false, fb0.v,
                                                  (short)0, a00, false, false);
    a01 = __builtin_amdgcn_wmma_f32_16x16x32_bf16(false, fa0.v, false, fb1.v,
                                                  (short)0, a01, false, false);
    a10 = __builtin_amdgcn_wmma_f32_16x16x32_bf16(false, fa1.v, false, fb0.v,
                                                  (short)0, a10, false, false);
    a11 = __builtin_amdgcn_wmma_f32_16x16x32_bf16(false, fa1.v, false, fb1.v,
                                                  (short)0, a11, false, false);
    __syncthreads();
  }

  // ---- store: lane = col (lid), vgpr v = row (half*8 + v) ----
#pragma unroll
  for (int j = 0; j < 2; ++j) {
    const int col = mBase + wc * 32 + j * 16 + lid;
    if (CM && col >= M) continue;
    const float bv = (bias != nullptr) ? bias[col] : 0.f;
#pragma unroll
    for (int i = 0; i < 2; ++i) {
      const v8f acc = (i == 0) ? (j == 0 ? a00 : a01) : (j == 0 ? a10 : a11);
      const int rowT = rowBase + wr * 32 + i * 16 + half * 8;
#pragma unroll
      for (int v = 0; v < 8; ++v) {
        float r = acc[v] + bv;
        if (ACT == 1)  // fast softplus: v_exp_f32 + v_log_f32
          r = (r > 20.f) ? r : __logf(1.f + __expf(r));
        float* p = C + (size_t)(rowT + v) * M + col;
        if (ACC) *p += r; else *p = r;
      }
    }
  }
}

// ---------------------------------------------------------------------------
// Depthwise causal conv (taps=4) + bias + SiLU.  reverse=1: anti-causal with
// flipped taps (== conv on reversed sequence, written un-reversed).
// ---------------------------------------------------------------------------
__global__ __launch_bounds__(256) void conv_silu_kernel(
    const float* __restrict__ xz,   // [B,L,2*DI]
    const float* __restrict__ cw,   // [DI,1,4]
    const float* __restrict__ cb,   // [DI]
    float* __restrict__ xc,         // [B,L,DI]
    int reverse) {
  const int idx = blockIdx.x * blockDim.x + threadIdx.x;
  if (idx >= BATCH * DI) return;
  const int b = idx / DI;
  const int d = idx - b * DI;
  const float w0 = cw[d * 4 + 0], w1 = cw[d * 4 + 1];
  const float w2 = cw[d * 4 + 2], w3 = cw[d * 4 + 3];
  const float bd = cb[d];
  const size_t base = ((size_t)b * SEQL) * (2 * DI) + d;
  if (!reverse) {
    float p1 = 0.f, p2 = 0.f, p3 = 0.f;  // x[l-1], x[l-2], x[l-3]
    for (int l = 0; l < SEQL; ++l) {
      const float cur = xz[base + (size_t)l * (2 * DI)];
      const float v = w3 * cur + w2 * p1 + w1 * p2 + w0 * p3 + bd;
      xc[((size_t)b * SEQL + l) * DI + d] = v * sigmoidf_(v);
      p3 = p2; p2 = p1; p1 = cur;
    }
  } else {
    float n1 = 0.f, n2 = 0.f, n3 = 0.f;  // x[l+1], x[l+2], x[l+3]
    for (int l = SEQL - 1; l >= 0; --l) {
      const float cur = xz[base + (size_t)l * (2 * DI)];
      const float v = w3 * cur + w2 * n1 + w1 * n2 + w0 * n3 + bd;
      xc[((size_t)b * SEQL + l) * DI + d] = v * sigmoidf_(v);
      n3 = n2; n2 = n1; n1 = cur;
    }
  }
}

// ---------------------------------------------------------------------------
// Selective scan + skip + gate. One thread per (b,d); 16-state in registers.
// ---------------------------------------------------------------------------
__global__ __launch_bounds__(256) void scan_kernel(
    const float* __restrict__ dt,     // [B,L,DI] (post softplus)
    const float* __restrict__ xdbl,   // [B,L,48]: [16:32)=B, [32:48)=C
    const float* __restrict__ xz,     // [B,L,2*DI] (z at +DI)
    float* __restrict__ xcio,         // in: conv out; out: y (in place)
    const float* __restrict__ A_log,  // [DI,16]
    const float* __restrict__ Dv,     // [DI]
    int reverse) {
  const int idx = blockIdx.x * blockDim.x + threadIdx.x;
  if (idx >= BATCH * DI) return;
  const int b = idx / DI;
  const int d = idx - b * DI;

  float A[DST];
#pragma unroll
  for (int s = 0; s < DST; ++s) A[s] = -__expf(A_log[d * DST + s]);
  const float Dd = Dv[d];
  float h[DST];
#pragma unroll
  for (int s = 0; s < DST; ++s) h[s] = 0.f;

  for (int t = 0; t < SEQL; ++t) {
    const int l = reverse ? (SEQL - 1 - t) : t;
    const size_t tok = (size_t)b * SEQL + l;
    const float dtv = dt[tok * DI + d];
    const float xv  = xcio[tok * DI + d];
    const float* Bp = xdbl + tok * XDBLW + DTR;
    const float* Cp = Bp + DST;
    const float dtx = dtv * xv;
    float y = 0.f;
#pragma unroll
    for (int s = 0; s < DST; ++s) {
      const float dA = __expf(dtv * A[s]);
      h[s] = dA * h[s] + dtx * Bp[s];
      y += h[s] * Cp[s];
    }
    const float zv = xz[tok * (2 * DI) + DI + d];
    xcio[tok * DI + d] = (y + xv * Dd) * (zv * sigmoidf_(zv));
  }
}

// ---------------------------------------------------------------------------
// Mean-pool over L, LayerNorm, SiLU, head GEMV. One block per batch element.
// ---------------------------------------------------------------------------
__global__ __launch_bounds__(256) void finalize_kernel(
    const float* __restrict__ dirout,  // [B,L,HID] (fwd+rev summed)
    const float* __restrict__ ln_g,
    const float* __restrict__ ln_b,
    const float* __restrict__ head_w,  // [HID,HID]
    const float* __restrict__ head_b,
    float* __restrict__ out) {
  __shared__ float s1[HID];
  __shared__ float s2[HID];
  const int b = blockIdx.x;
  const int h = threadIdx.x;

  float sum = 0.f;
  for (int l = 0; l < SEQL; ++l)
    sum += dirout[((size_t)b * SEQL + l) * HID + h];
  const float pooled = sum * (1.f / (float)SEQL);

  s1[h] = pooled;
  __syncthreads();
  for (int off = 128; off > 0; off >>= 1) {
    if (h < off) s1[h] += s1[h + off];
    __syncthreads();
  }
  const float mu = s1[0] * (1.f / (float)HID);
  __syncthreads();
  const float dev = pooled - mu;
  s1[h] = dev * dev;
  __syncthreads();
  for (int off = 128; off > 0; off >>= 1) {
    if (h < off) s1[h] += s1[h + off];
    __syncthreads();
  }
  const float var = s1[0] * (1.f / (float)HID);
  __syncthreads();

  const float hn = dev * rsqrtf(var + 1e-5f) * ln_g[h] + ln_b[h];
  s2[h] = hn * sigmoidf_(hn);  // silu
  __syncthreads();

  float a = head_b[h];
  for (int k = 0; k < HID; ++k) a += s2[k] * head_w[(size_t)h * HID + k];
  out[(size_t)b * HID + h] = a;
}

// ---------------------------------------------------------------------------
// Host orchestration
// ---------------------------------------------------------------------------
template <bool CK, bool CM, int ACT, bool ACC>
static inline void launch_gemm(const float* X, int ldx, const float* W,
                               const float* bias, float* C, int N, int K, int M,
                               hipStream_t stream) {
  dim3 grid(N / 64, (M + 127) / 128);
  gemm_bf16_wmma_kernel<CK, CM, ACT, ACC><<<grid, dim3(256), 0, stream>>>(
      X, ldx, W, bias, C, N, K, M);
}

extern "C" void kernel_launch(void* const* d_in, const int* in_sizes, int n_in,
                              void* d_out, int out_size, void* d_ws,
                              size_t ws_size, hipStream_t stream) {
  (void)in_sizes; (void)n_in; (void)out_size; (void)ws_size;

  const float* text_seq  = (const float*)d_in[0];   // [B,L,768]
  const float* proj_in_w = (const float*)d_in[1];   // [256,768]
  const float* proj_in_b = (const float*)d_in[2];   // [256]
  const float* ln_g      = (const float*)d_in[21];
  const float* ln_b      = (const float*)d_in[22];
  const float* head_w    = (const float*)d_in[23];
  const float* head_b    = (const float*)d_in[24];
  float* out = (float*)d_out;

  // workspace layout (floats)
  float* ws      = (float*)d_ws;
  float* xbuf    = ws;                                   // [NTOK, HID]
  float* xzbuf   = xbuf    + (size_t)NTOK * HID;         // [NTOK, 2*DI]
  float* xcbuf   = xzbuf   + (size_t)NTOK * 2 * DI;      // [NTOK, DI]
  float* xdblbuf = xcbuf   + (size_t)NTOK * DI;          // [NTOK, 48]
  float* dtbuf   = xdblbuf + (size_t)NTOK * XDBLW;       // [NTOK, DI]
  float* dobuf   = dtbuf   + (size_t)NTOK * DI;          // [NTOK, HID]

  const int ew = (BATCH * DI) / 256;  // 512 elementwise blocks

  // 1) input projection: x = text_seq @ proj_in_w^T + b   (K=768, M=256)
  launch_gemm<false, false, 0, false>(text_seq, DMODEL, proj_in_w, proj_in_b,
                                      xbuf, NTOK, DMODEL, HID, stream);

  for (int dir = 0; dir < 2; ++dir) {
    const int base = 3 + dir * 9;
    const float* in_w    = (const float*)d_in[base + 0];  // [1024,256]
    const float* conv_w  = (const float*)d_in[base + 1];  // [512,1,4]
    const float* conv_b  = (const float*)d_in[base + 2];  // [512]
    const float* xproj_w = (const float*)d_in[base + 3];  // [48,512]
    const float* dt_w    = (const float*)d_in[base + 4];  // [512,16]
    const float* dt_b    = (const float*)d_in[base + 5];  // [512]
    const float* A_log   = (const float*)d_in[base + 6];  // [512,16]
    const float* Dvec    = (const float*)d_in[base + 7];  // [512]
    const float* out_w   = (const float*)d_in[base + 8];  // [256,512]

    // xz = x @ in_w^T          (K=256, M=1024)
    launch_gemm<false, false, 0, false>(xbuf, HID, in_w, nullptr, xzbuf,
                                        NTOK, HID, 2 * DI, stream);
    // depthwise conv + SiLU (direction-aware)
    conv_silu_kernel<<<ew, 256, 0, stream>>>(xzbuf, conv_w, conv_b, xcbuf, dir);
    // x_dbl = xc @ xproj_w^T   (K=512, M=48 -> column masking)
    launch_gemm<false, true, 0, false>(xcbuf, DI, xproj_w, nullptr, xdblbuf,
                                       NTOK, DI, XDBLW, stream);
    // dt = softplus(x_dbl[:, :16] @ dt_w^T + dt_b)   (K=16 -> zero-pad)
    launch_gemm<true, false, 1, false>(xdblbuf, XDBLW, dt_w, dt_b, dtbuf,
                                       NTOK, DTR, DI, stream);
    // selective scan + skip + gate (direction-aware), y in place over xcbuf
    scan_kernel<<<ew, 256, 0, stream>>>(dtbuf, xdblbuf, xzbuf, xcbuf, A_log,
                                        Dvec, dir);
    // out = y @ out_w^T        (K=512, M=256); accumulate reverse onto forward
    if (dir == 0)
      launch_gemm<false, false, 0, false>(xcbuf, DI, out_w, nullptr, dobuf,
                                          NTOK, DI, HID, stream);
    else
      launch_gemm<false, false, 0, true>(xcbuf, DI, out_w, nullptr, dobuf,
                                         NTOK, DI, HID, stream);
  }

  // pool + layernorm + silu + head
  finalize_kernel<<<BATCH, 256, 0, stream>>>(dobuf, ln_g, ln_b, head_w, head_b,
                                             out);
}
